// GCNLSTMBaseline_38543036514576
// MI455X (gfx1250) — matrix operand order repaired and checked by reading, not compiled, not run on hardware
//
#include <hip/hip_runtime.h>
#include <hip/hip_bf16.h>

#define TT    12
#define FIN   32
#define HH    64
#define NCOLS 256   // 4 gates * 64
#define KTOT  288   // 3*32 + 3*64

typedef __attribute__((ext_vector_type(16))) __bf16 v16bf;
typedef __attribute__((ext_vector_type(8)))  __bf16 v8bf;
typedef __attribute__((ext_vector_type(8)))  float  v8f;

__device__ inline __bf16 f2bf(float f) {
  unsigned u = __builtin_bit_cast(unsigned, f);
  u += 0x7FFFu + ((u >> 16) & 1u);            // round-to-nearest-even
  unsigned short hs = (unsigned short)(u >> 16);
  return __builtin_bit_cast(__bf16, hs);
}

__global__ void zero_kernel(float* __restrict__ p, long n) {
  long i = (long)blockIdx.x * blockDim.x + threadIdx.x;
  if (i < n) p[i] = 0.0f;
}

__global__ void neg_copy_kernel(float* __restrict__ dst, const float* __restrict__ src, long n) {
  long i = (long)blockIdx.x * blockDim.x + threadIdx.x;
  if (i < n) dst[i] = -src[i];
}

__global__ void deg_kernel(const int* __restrict__ row, const float* __restrict__ ew,
                           float* __restrict__ deg, int E) {
  int e = blockIdx.x * blockDim.x + threadIdx.x;
  if (e < E) atomicAdd(deg + row[e], ew[e]);
}

__global__ void norm_kernel(const int* __restrict__ row, const int* __restrict__ col,
                            const float* __restrict__ ew, const float* __restrict__ deg,
                            float* __restrict__ nrm, int E) {
  int e = blockIdx.x * blockDim.x + threadIdx.x;
  if (e >= E) return;
  float dr = deg[row[e]];
  float dc = deg[col[e]];
  float ir = dr > 0.0f ? rsqrtf(fmaxf(dr, 1e-12f)) : 0.0f;
  float ic = dc > 0.0f ? rsqrtf(fmaxf(dc, 1e-12f)) : 0.0f;
  nrm[e] = -ir * ew[e] * ic;
}

// out[col[e], :] += scale * nrm[e] * z[row[e], :]  (C channels, 4 per thread)
__global__ void prop_kernel(const float4* __restrict__ z, const int* __restrict__ row,
                            const int* __restrict__ col, const float* __restrict__ nrm,
                            float* __restrict__ out, long total, int qshift, float scale) {
  long idx = (long)blockIdx.x * blockDim.x + threadIdx.x;
  if (idx >= total) return;
  long e = idx >> qshift;
  int  q = (int)(idx & ((1 << qshift) - 1));
  float w = scale * nrm[e];
  if (w == 0.0f) return;
  float4 v = z[((long)row[e] << qshift) + q];
  float* dst = out + ((((long)col[e] << qshift) + q) << 2);
  atomicAdd(dst + 0, w * v.x);
  atomicAdd(dst + 1, w * v.y);
  atomicAdd(dst + 2, w * v.z);
  atomicAdd(dst + 3, w * v.w);
}

// One-time: pack all gate weights to bf16, TRANSPOSED [col 0..255][k 0..287].
// col = gate*64 + n. k<96 -> Wx(gate) flat [k*64+n]; else Wh(gate) flat [(k-96)*64+n].
__global__ void pack_weights_kernel(
    const float* __restrict__ Wxi, const float* __restrict__ Wxf,
    const float* __restrict__ Wxc, const float* __restrict__ Wxo,
    const float* __restrict__ Whi, const float* __restrict__ Whf,
    const float* __restrict__ Whc, const float* __restrict__ Who,
    __bf16* __restrict__ Bpack) {
  int idx = blockIdx.x * blockDim.x + threadIdx.x;   // col*288 + k
  if (idx >= NCOLS * KTOT) return;
  int col = idx / KTOT;
  int k   = idx - col * KTOT;
  int g    = col >> 6;
  int ncol = col & 63;
  float v;
  if (k < 96) {
    const float* W = (g == 0) ? Wxi : (g == 1) ? Wxf : (g == 2) ? Wxc : Wxo;
    v = W[(long)k * 64 + ncol];
  } else {
    int kh = k - 96;
    const float* W = (g == 0) ? Whi : (g == 1) ? Whf : (g == 2) ? Whc : Who;
    v = W[(long)kh * 64 + ncol];
  }
  Bpack[idx] = f2bf(v);
}

__device__ inline const float* a_src(int node, int kg,
    const float* xt, const float* Tx1, const float* Tx2,
    const float* h, const float* Th1, const float* Th2) {
  if (kg < 96) {
    const float* s = (kg < 32) ? xt : (kg < 64 ? Tx1 : Tx2);
    return s + (long)node * 32 + (kg & 31);
  }
  int kh = kg - 96;
  const float* s = (kh < 64) ? h : (kh < 128 ? Th1 : Th2);
  return s + (long)node * 64 + (kh & 63);
}

// G[N,256] = [x|Tx1|Tx2|h|Th1|Th2] (N x 288) @ Bpack^T (288 x 256), bf16 WMMA, f32 acc
// Block tile: 128 rows x 256 cols. 8 waves, each wave: 16 rows x 256 cols (16 WMMA n-tiles).
__global__ __launch_bounds__(256) void gate_gemm_kernel(
    const float* __restrict__ xt,  const float* __restrict__ Tx1, const float* __restrict__ Tx2,
    const float* __restrict__ h,   const float* __restrict__ Th1, const float* __restrict__ Th2,
    const __bf16* __restrict__ Bpack,
    float* __restrict__ G, int Nn) {
  __shared__ __bf16 As[128][40];   // [row][k], row stride 80B (16B multiple, 20-word bank stride)
  __shared__ __bf16 Bs[256][40];   // TRANSPOSED: [col][k], same stride

  const int tid  = threadIdx.x;
  const int lane = tid & 31;
  const int wy   = tid >> 5;          // wave id 0..7 -> 16-row slice
  const int l16  = lane & 15;
  const int lh   = lane >> 4;         // 0/1
  const int rowBase = blockIdx.x * 128;

  v8f acc[16] = {};

  const int ar  = tid >> 1;           // A row this thread stages (0..127)
  const int ak0 = (tid & 1) * 16;     // A k base (0 or 16)

  for (int kk = 0; kk < 9; ++kk) {
    const int kgBase = kk * 32;
    // ---- stage A tile (128 x 32): 16 elems/thread, packed v8bf stores ----
    {
      int node = rowBase + ar;
      v8bf av0, av1;
      #pragma unroll
      for (int i = 0; i < 16; ++i) {
        int kg = kgBase + ak0 + i;
        float v = 0.0f;
        if (node < Nn) v = *a_src(node, kg, xt, Tx1, Tx2, h, Th1, Th2);
        if (i < 8) av0[i] = f2bf(v);
        else       av1[i - 8] = f2bf(v);
      }
      *reinterpret_cast<v8bf*>(&As[ar][ak0])     = av0;
      *reinterpret_cast<v8bf*>(&As[ar][ak0 + 8]) = av1;
      // prefetch next k-strip of A sources (global_prefetch_b8)
      if (node < Nn && kgBase + 32 < KTOT)
        __builtin_prefetch(a_src(node, kgBase + ak0 + 32, xt, Tx1, Tx2, h, Th1, Th2), 0, 3);
    }
    // ---- stage B strip (32 x 256, pre-packed bf16, pre-transposed): pure 64B copy ----
    {
      const v8bf* src = reinterpret_cast<const v8bf*>(Bpack + (long)tid * KTOT + kgBase);
      #pragma unroll
      for (int c4 = 0; c4 < 4; ++c4)
        *reinterpret_cast<v8bf*>(&Bs[tid][c4 * 8]) = src[c4];
    }
    __syncthreads();
    // ---- A fragment: two aligned 16B LDS loads + shuffle (per documented layout) ----
    v8bf a0 = *reinterpret_cast<const v8bf*>(&As[wy * 16 + l16][lh * 8]);
    v8bf a1 = *reinterpret_cast<const v8bf*>(&As[wy * 16 + l16][16 + lh * 8]);
    v16bf a = __builtin_shufflevector(a0, a1, 0,1,2,3,4,5,6,7,8,9,10,11,12,13,14,15);
    #pragma unroll
    for (int nt = 0; nt < 16; ++nt) {
      int colb = nt * 16 + l16;
      v8bf b0 = *reinterpret_cast<const v8bf*>(&Bs[colb][lh * 8]);
      v8bf b1 = *reinterpret_cast<const v8bf*>(&Bs[colb][16 + lh * 8]);
      v16bf b = __builtin_shufflevector(b0, b1, 0,1,2,3,4,5,6,7,8,9,10,11,12,13,14,15);
      acc[nt] = __builtin_amdgcn_wmma_f32_16x16x32_bf16(
          false, a, false, b, (short)0, acc[nt], false, false);
    }
    __syncthreads();
  }
  // ---- epilogue: C/D 16x16 f32 layout ----
  #pragma unroll
  for (int nt = 0; nt < 16; ++nt) {
    int colg = nt * 16 + l16;
    #pragma unroll
    for (int v = 0; v < 8; ++v) {
      int r = rowBase + wy * 16 + lh * 8 + v;
      if (r < Nn) G[(long)r * NCOLS + colg] = acc[nt][v];
    }
  }
}

__global__ void lstm_kernel(const float* __restrict__ G, float* __restrict__ h, float* __restrict__ c,
    const float* __restrict__ bxi, const float* __restrict__ bhi, const float* __restrict__ wci, const float* __restrict__ bi,
    const float* __restrict__ bxf, const float* __restrict__ bhf, const float* __restrict__ wcf, const float* __restrict__ bf,
    const float* __restrict__ bxc, const float* __restrict__ bhc, const float* __restrict__ bc,
    const float* __restrict__ bxo, const float* __restrict__ bho, const float* __restrict__ wco, const float* __restrict__ bo,
    long n) {
  long idx = (long)blockIdx.x * blockDim.x + threadIdx.x;
  if (idx >= n) return;
  int  k    = (int)(idx & (HH - 1));
  long base = (idx >> 6) * NCOLS;
  float cv = c[idx];
  float gi = G[base + k]        + bxi[k] + bhi[k] + wci[k] * cv + bi[k];
  float gf = G[base + 64 + k]   + bxf[k] + bhf[k] + wcf[k] * cv + bf[k];
  float gc = G[base + 128 + k]  + bxc[k] + bhc[k] + bc[k];
  float go = G[base + 192 + k]  + bxo[k] + bho[k] + bo[k];
  float iv = 1.0f / (1.0f + __expf(-gi));
  float fv = 1.0f / (1.0f + __expf(-gf));
  float tv = tanhf(gc);
  float cn = fv * cv + iv * tv;
  float ov = 1.0f / (1.0f + __expf(-(go + wco[k] * cn)));
  c[idx] = cn;
  h[idx] = ov * tanhf(cn);
}

__global__ void head_kernel(const float* __restrict__ h, const float* __restrict__ Wl,
                            const float* __restrict__ bl, float* __restrict__ out, int N) {
  int n = blockIdx.x * blockDim.x + threadIdx.x;
  if (n >= N) return;
  float s = bl[0];
  #pragma unroll
  for (int k = 0; k < HH; ++k) s += h[(long)n * HH + k] * Wl[k];
  out[n] = s;
}

__global__ void copy_hc_kernel(const float* __restrict__ h, const float* __restrict__ c,
                               float* __restrict__ out, long nh) {
  long idx = (long)blockIdx.x * blockDim.x + threadIdx.x;
  if (idx >= 2 * nh) return;
  out[idx] = (idx < nh) ? h[idx] : c[idx - nh];
}

static inline unsigned gblocks(long n, int tb) { return (unsigned)((n + tb - 1) / tb); }

extern "C" void kernel_launch(void* const* d_in, const int* in_sizes, int n_in,
                              void* d_out, int out_size, void* d_ws, size_t ws_size,
                              hipStream_t stream) {
  const float* x    = (const float*)d_in[0];
  const int*   eidx = (const int*)  d_in[1];
  const float* ewal = (const float*)d_in[2];
  const float* Wxi = (const float*)d_in[3];  const float* bxi = (const float*)d_in[4];
  const float* Whi = (const float*)d_in[5];  const float* bhi = (const float*)d_in[6];
  const float* wci = (const float*)d_in[7];  const float* bi  = (const float*)d_in[8];
  const float* Wxf = (const float*)d_in[9];  const float* bxf = (const float*)d_in[10];
  const float* Whf = (const float*)d_in[11]; const float* bhf = (const float*)d_in[12];
  const float* wcf = (const float*)d_in[13]; const float* bf  = (const float*)d_in[14];
  const float* Wxc = (const float*)d_in[15]; const float* bxc = (const float*)d_in[16];
  const float* Whc = (const float*)d_in[17]; const float* bhc = (const float*)d_in[18];
  const float* bc  = (const float*)d_in[19];
  const float* Wxo = (const float*)d_in[20]; const float* bxo = (const float*)d_in[21];
  const float* Who = (const float*)d_in[22]; const float* bho = (const float*)d_in[23];
  const float* wco = (const float*)d_in[24]; const float* bo  = (const float*)d_in[25];
  const float* Wl  = (const float*)d_in[26]; const float* bl  = (const float*)d_in[27];

  const int N = in_sizes[0] / (TT * FIN);
  const int E = in_sizes[2] / TT;
  const int TB = 256;

  float* ws = (float*)d_ws;
  size_t off = 0;
  float* deg = ws + off; off += (size_t)N;
  float* nrm = ws + off; off += (size_t)E;
  float* Tx1 = ws + off; off += (size_t)N * FIN;
  float* Tx2 = ws + off; off += (size_t)N * FIN;
  float* Th1 = ws + off; off += (size_t)N * HH;
  float* Th2 = ws + off; off += (size_t)N * HH;
  float* hb  = ws + off; off += (size_t)N * HH;
  float* cb  = ws + off; off += (size_t)N * HH;   // contiguous with hb
  float* G   = ws + off; off += (size_t)N * NCOLS;
  __bf16* Bpack = (__bf16*)(ws + off); off += (size_t)(NCOLS * KTOT) / 2;  // 144KB bf16

  // h = c = 0 every call (deterministic); pack weights once per launch
  zero_kernel<<<gblocks((long)N * HH * 2, TB), TB, 0, stream>>>(hb, (long)N * HH * 2);
  pack_weights_kernel<<<gblocks(NCOLS * KTOT, TB), TB, 0, stream>>>(
      Wxi, Wxf, Wxc, Wxo, Whi, Whf, Whc, Who, Bpack);

  for (int t = 0; t < TT; ++t) {
    const int*   row = eidx + (size_t)t * 2 * E;
    const int*   col = row + E;
    const float* ew  = ewal + (size_t)t * E;
    const float* xt  = x + (size_t)t * N * FIN;

    zero_kernel<<<gblocks(N, TB), TB, 0, stream>>>(deg, N);
    deg_kernel<<<gblocks(E, TB), TB, 0, stream>>>(row, ew, deg, E);
    norm_kernel<<<gblocks(E, TB), TB, 0, stream>>>(row, col, ew, deg, nrm, E);

    // Tx1 = prop(x)
    zero_kernel<<<gblocks((long)N * FIN, TB), TB, 0, stream>>>(Tx1, (long)N * FIN);
    prop_kernel<<<gblocks((long)E * (FIN / 4), TB), TB, 0, stream>>>(
        (const float4*)xt, row, col, nrm, Tx1, (long)E * (FIN / 4), 3, 1.0f);
    // Tx2 = 2*prop(Tx1) - x
    neg_copy_kernel<<<gblocks((long)N * FIN, TB), TB, 0, stream>>>(Tx2, xt, (long)N * FIN);
    prop_kernel<<<gblocks((long)E * (FIN / 4), TB), TB, 0, stream>>>(
        (const float4*)Tx1, row, col, nrm, Tx2, (long)E * (FIN / 4), 3, 2.0f);

    // Th1 = prop(h)
    zero_kernel<<<gblocks((long)N * HH, TB), TB, 0, stream>>>(Th1, (long)N * HH);
    prop_kernel<<<gblocks((long)E * (HH / 4), TB), TB, 0, stream>>>(
        (const float4*)hb, row, col, nrm, Th1, (long)E * (HH / 4), 4, 1.0f);
    // Th2 = 2*prop(Th1) - h
    neg_copy_kernel<<<gblocks((long)N * HH, TB), TB, 0, stream>>>(Th2, hb, (long)N * HH);
    prop_kernel<<<gblocks((long)E * (HH / 4), TB), TB, 0, stream>>>(
        (const float4*)Th1, row, col, nrm, Th2, (long)E * (HH / 4), 4, 2.0f);

    gate_gemm_kernel<<<gblocks(N, 128), 256, 0, stream>>>(
        xt, Tx1, Tx2, hb, Th1, Th2, Bpack, G, N);

    lstm_kernel<<<gblocks((long)N * HH, TB), TB, 0, stream>>>(
        G, hb, cb, bxi, bhi, wci, bi, bxf, bhf, wcf, bf,
        bxc, bhc, bc, bxo, bho, wco, bo, (long)N * HH);
  }

  float* out = (float*)d_out;
  head_kernel<<<gblocks(N, TB), TB, 0, stream>>>(hb, Wl, bl, out, N);
  copy_hc_kernel<<<gblocks((long)2 * N * HH, TB), TB, 0, stream>>>(hb, cb, out + N, (long)N * HH);

  (void)n_in; (void)out_size; (void)ws_size;
}
